// SpGraphAttentionLayer_90280212562307
// MI455X (gfx1250) — compile-verified, matrix operand order checked by reading, not run on hardware
//
#include <hip/hip_runtime.h>
#include <hip/hip_bf16.h>

// ---- problem constants (from reference) ----
#define N_NODES 50000
#define IN_F    200
#define OUT_F   200
#define NRELA   200
#define E1      400000
#define E2      100000
#define E_TOT   500000
#define KDIM    600        // 2*IN_F + NRELA
#define NSTEPS  19         // ceil(600/32) -> K padded to 608
#define NT      13         // ceil(200/16) -> N padded to 208
#define ROWP    616        // padded LDS row length (bf16 elems), conflict-free stride
#define EM_STR  212        // padded f32 edge_m row stride in LDS
#define ALPHA   0.2f

typedef __attribute__((ext_vector_type(16))) __bf16 v16bf;
typedef __attribute__((ext_vector_type(8)))  __bf16 v8bf;
typedef __attribute__((ext_vector_type(8)))  float  v8f;
typedef __attribute__((ext_vector_type(4)))  float  v4f;

__device__ __forceinline__ unsigned short f2bf(float f) {
  unsigned u = __float_as_uint(f);
  u = (u + 0x7FFFu + ((u >> 16) & 1u)) >> 16;   // RNE f32 -> bf16
  return (unsigned short)u;
}

// ---------------------------------------------------------------------------
// Pack a[OUT_F, KDIM] (f32) into per-lane WMMA B-fragment order (bf16):
// layout index = ((s*NT + t)*32 + lane)*16 + e
// ---------------------------------------------------------------------------
__global__ void pack_weights_kernel(const float* __restrict__ a,
                                    unsigned short* __restrict__ aPack) {
  int i = blockIdx.x * blockDim.x + threadIdx.x;
  const int total = NSTEPS * NT * 32 * 16;
  if (i >= total) return;
  int e = i & 15;
  int l = (i >> 4) & 31;
  int t = (i >> 9) % NT;
  int s = (i >> 9) / NT;
  int j = e >> 1, sub = e & 1;
  int base = (l < 16) ? 0 : 8;
  int kl = (j < 4) ? (2 * j + sub + base) : (16 + 2 * (j - 4) + sub + base);
  int kg = s * 32 + kl;
  int col = t * 16 + (l & 15);
  float v = (col < OUT_F && kg < KDIM) ? a[col * KDIM + kg] : 0.0f;
  aPack[i] = f2bf(v);
}

// ---------------------------------------------------------------------------
// Main edge kernel: ONE wave (32 threads) per workgroup, 32 edges per WG.
// The wave computes TWO 16-edge M-tiles x 13 N-tiles; each B fragment is
// loaded once per K-step and reused for both M-tiles (halves L2 B-traffic).
// ---------------------------------------------------------------------------
__launch_bounds__(32)
__global__ void gat_edge_kernel(
    const float* __restrict__ x,
    const int*   __restrict__ edge,            // [2, E1] flat
    const float* __restrict__ edge_embed,      // [E1, NRELA]
    const int*   __restrict__ edge_nhop,       // [2, E2] flat
    const float* __restrict__ edge_embed_nhop, // [E2, NRELA]
    const float* __restrict__ a2,              // [OUT_F]
    const float* __restrict__ mlp_w,           // [OUT_F]
    const float* __restrict__ mlp_b,           // [1]
    const unsigned short* __restrict__ aPack,  // packed bf16 B fragments
    float* __restrict__ rowsum,                // [N_NODES]
    float* __restrict__ hprime)                // [N_NODES, OUT_F] == d_out
{
  __shared__ __align__(16) unsigned short sEdgeH[32 * ROWP]; // bf16 tile, reused as f32 edge_m
  __shared__ float sEE[32];
  __shared__ int   sSrc[32];

  const int lane  = threadIdx.x;
  const int eBase = blockIdx.x * 32;

  // ---- Phase 1: gather + convert edge_h tile into LDS (bf16, zero-padded) ----
  for (int idx = lane; idx < 32 * (ROWP / 4); idx += 32) {
    int row = idx / (ROWP / 4);
    int q   = idx % (ROWP / 4);           // quad of 4 elements
    int g = eBase + row;
    unsigned short* dptr = sEdgeH + row * ROWP + q * 4;
    ushort4 o;
    if (g < E_TOT && q < 150) {
      int si, di; const float* embp;
      if (g < E1) {
        si = edge[g]; di = edge[E1 + g];
        embp = edge_embed + (size_t)g * NRELA;
      } else {
        int g2 = g - E1;
        si = edge_nhop[g2]; di = edge_nhop[E2 + g2];
        embp = edge_embed_nhop + (size_t)g2 * NRELA;
      }
      v4f v;
      if (q < 50)       v = *(const v4f*)(x + (size_t)si * IN_F + q * 4);
      else if (q < 100) v = *(const v4f*)(x + (size_t)di * IN_F + (q - 50) * 4);
      else              v = __builtin_nontemporal_load(
                              (const v4f*)(embp + (q - 100) * 4)); // streamed once
      o = make_ushort4(f2bf(v.x), f2bf(v.y), f2bf(v.z), f2bf(v.w));
    } else {
      o = make_ushort4(0, 0, 0, 0);
    }
    *(ushort4*)dptr = o;
  }
  __syncthreads();

  const int m16  = lane & 15;
  const int half = lane >> 4;

  v8f acc0[NT], acc1[NT];
#pragma unroll
  for (int t = 0; t < NT; ++t) {
    acc0[t] = (v8f){0.f, 0.f, 0.f, 0.f, 0.f, 0.f, 0.f, 0.f};
    acc1[t] = (v8f){0.f, 0.f, 0.f, 0.f, 0.f, 0.f, 0.f, 0.f};
  }

  const unsigned short* rowp0 = sEdgeH + m16 * ROWP;          // M-tile 0: rows 0..15
  const unsigned short* rowp1 = sEdgeH + (16 + m16) * ROWP;   // M-tile 1: rows 16..31
  const v16bf* bptr = (const v16bf*)aPack;

#pragma unroll 1
  for (int s = 0; s < NSTEPS; ++s) {
    const int k0 = s * 32 + half * 8;
    v8bf lo0 = *(const v8bf*)(rowp0 + k0);
    v8bf hi0 = *(const v8bf*)(rowp0 + k0 + 16);
    v8bf lo1 = *(const v8bf*)(rowp1 + k0);
    v8bf hi1 = *(const v8bf*)(rowp1 + k0 + 16);
    v16bf a0 = __builtin_shufflevector(lo0, hi0,
        0, 1, 2, 3, 4, 5, 6, 7, 8, 9, 10, 11, 12, 13, 14, 15);
    v16bf a1 = __builtin_shufflevector(lo1, hi1,
        0, 1, 2, 3, 4, 5, 6, 7, 8, 9, 10, 11, 12, 13, 14, 15);
#pragma unroll
    for (int t = 0; t < NT; ++t) {
      v16bf bfrag = bptr[(s * NT + t) * 32 + lane];
      acc0[t] = __builtin_amdgcn_wmma_f32_16x16x32_bf16(
          false, a0, false, bfrag, (short)0, acc0[t], false, false);
      acc1[t] = __builtin_amdgcn_wmma_f32_16x16x32_bf16(
          false, a1, false, bfrag, (short)0, acc1[t], false, false);
    }
  }

  __syncthreads();  // edge_h tile dead; reuse LDS as f32 edge_m staging (32 x EM_STR)

  float* em = (float*)sEdgeH;
#pragma unroll
  for (int t = 0; t < NT; ++t) {
    int col = t * 16 + m16;
#pragma unroll
    for (int r = 0; r < 8; ++r) {
      int rowm = half ? (8 + r) : r;   // C/D layout: lanes 0-15 hold M=r, 16-31 M=8+r
      em[rowm * EM_STR + col]        = acc0[t][r];
      em[(16 + rowm) * EM_STR + col] = acc1[t][r];
    }
  }

  // ---- per-edge dots: lane pair (l, l+16) splits columns mod 2; 2 subtiles ----
#pragma unroll 1
  for (int u = 0; u < 2; ++u) {
    float s1 = 0.f, s2 = 0.f;
    const float* emrow = em + (u * 16 + m16) * EM_STR;
    for (int c = half; c < OUT_F; c += 2) {
      float v = emrow[c];
      s1 += v * a2[c];
      s2 += v * mlp_w[c];
    }
    s1 += __shfl_xor(s1, 16, 32);
    s2 += __shfl_xor(s2, 16, 32);

    float lr = (s1 > 0.f) ? s1 : ALPHA * s1;          // leaky_relu
    float y  = tanhf(s2 + mlp_b[0]);
    float p  = (-lr) * y * (1.0f / (float)E_TOT);
    float ee = expf(p);

    int g = eBase + u * 16 + m16;
    if (half == 0 && g < E_TOT) {
      int sidx = (g < E1) ? edge[g] : edge_nhop[g - E1];
      unsafeAtomicAdd(rowsum + sidx, ee);
      sEE[u * 16 + m16]  = ee;
      sSrc[u * 16 + m16] = sidx;
    }
  }

  // ---- weighted scatter of edge_m rows into h_prime (f32 global atomics) ----
  for (int m = 0; m < 32; ++m) {
    int ge = eBase + m;
    if (ge >= E_TOT) break;
    float w  = sEE[m];
    int   so = sSrc[m];
    float* op = hprime + (size_t)so * OUT_F;
    const float* er = em + m * EM_STR;
    for (int c = lane; c < OUT_F; c += 32)
      unsafeAtomicAdd(op + c, er[c] * w);
  }
}

// ---------------------------------------------------------------------------
// Finalize: h_prime / rowsum (0 -> 1e-12), then ELU, in place on d_out.
// ---------------------------------------------------------------------------
__global__ void finalize_kernel(float* __restrict__ out,
                                const float* __restrict__ rowsum) {
  int i = blockIdx.x * blockDim.x + threadIdx.x;
  if (i >= N_NODES * OUT_F) return;
  int n = i / OUT_F;
  float rs = rowsum[n];
  if (rs == 0.f) rs = 1e-12f;
  float v = out[i] / rs;
  out[i] = (v > 0.f) ? v : (expf(v) - 1.f);
}

extern "C" void kernel_launch(void* const* d_in, const int* in_sizes, int n_in,
                              void* d_out, int out_size, void* d_ws, size_t ws_size,
                              hipStream_t stream) {
  const float* x    = (const float*)d_in[0];
  const int*   edge = (const int*)  d_in[1];
  const float* ee1  = (const float*)d_in[2];
  const int*   en   = (const int*)  d_in[3];
  const float* ee2  = (const float*)d_in[4];
  const float* a    = (const float*)d_in[5];
  const float* a2   = (const float*)d_in[6];
  const float* mw   = (const float*)d_in[7];
  const float* mb   = (const float*)d_in[8];
  (void)in_sizes; (void)n_in; (void)out_size; (void)ws_size;

  unsigned short* aPack = (unsigned short*)d_ws;                 // 252,928 B
  float* rowsum = (float*)((char*)d_ws + 256 * 1024);            // 200,000 B
  float* out = (float*)d_out;

  hipMemsetAsync(out, 0, (size_t)N_NODES * OUT_F * sizeof(float), stream);
  hipMemsetAsync(rowsum, 0, (size_t)N_NODES * sizeof(float), stream);

  const int packTotal = NSTEPS * NT * 32 * 16;
  pack_weights_kernel<<<(packTotal + 255) / 256, 256, 0, stream>>>(a, aPack);

  gat_edge_kernel<<<(E_TOT + 31) / 32, 32, 0, stream>>>(
      x, edge, ee1, en, ee2, a2, mw, mb, aPack, rowsum, out);

  const int tot = N_NODES * OUT_F;
  finalize_kernel<<<(tot + 255) / 256, 256, 0, stream>>>(out, rowsum);
}
